// GCN_3l_norm_37787122270455
// MI455X (gfx1250) — matrix operand order repaired and checked by reading, not compiled, not run on hardware
//
#include <hip/hip_runtime.h>

#define N_NODES 100000
#define E_EDGES 1600000
#define F_IN    256
#define H_DIM   128
#define C_OUT   16
#define BN_EPS  1e-5f

typedef __bf16          v16bf __attribute__((ext_vector_type(16)));
typedef unsigned short  v16u  __attribute__((ext_vector_type(16)));
typedef unsigned short  u16x8 __attribute__((ext_vector_type(8)));
typedef float           v8f   __attribute__((ext_vector_type(8)));
typedef unsigned int    u32x4 __attribute__((ext_vector_type(4)));
typedef int             i32x8 __attribute__((ext_vector_type(8)));
typedef int             i32x4 __attribute__((ext_vector_type(4)));

__device__ __forceinline__ unsigned short f2bf(float f) {
    union { float f; unsigned int u; } v; v.f = f;
    unsigned int u = v.u;
    unsigned int r = u + 0x7FFFu + ((u >> 16) & 1u);   // round-to-nearest-even
    return (unsigned short)(r >> 16);
}

__device__ __forceinline__ void atomAddF32(float* p, float v) {
    __hip_atomic_fetch_add(p, v, __ATOMIC_RELAXED, __HIP_MEMORY_SCOPE_AGENT);
}

// ---------------- utility ----------------
__global__ void k_fill(float* p, float val, int n) {
    int i = blockIdx.x * blockDim.x + threadIdx.x;
    if (i < n) p[i] = val;
}

__global__ void k_cvt_bf16(const float* __restrict__ src, unsigned short* __restrict__ dst, int n) {
    int i = blockIdx.x * blockDim.x + threadIdx.x;
    if (i < n) dst[i] = f2bf(src[i]);
}

// src [K, H_DIM] row-major  ->  dst [H_DIM, K] (transposed, bf16)
__global__ void k_cvt_tr_bf16(const float* __restrict__ src, unsigned short* __restrict__ dst,
                              int K) {
    int i = blockIdx.x * blockDim.x + threadIdx.x;
    if (i < K * H_DIM) {
        int k = i >> 7, col = i & 127;
        dst[col * K + k] = f2bf(src[i]);
    }
}

// ---------------- gcn_norm ----------------
__global__ void k_deg(const int* __restrict__ ei, const float* __restrict__ ew,
                      float* __restrict__ deg) {
    int stride = gridDim.x * blockDim.x;
    for (int e = blockIdx.x * blockDim.x + threadIdx.x; e < E_EDGES; e += stride) {
        __builtin_prefetch(&ei[E_EDGES + e + stride], 0, 0);
        int c = ei[E_EDGES + e];
        atomAddF32(&deg[c], ew[e]);
    }
}

__global__ void k_dinv(const float* __restrict__ deg, float* __restrict__ dinv) {
    int n = blockIdx.x * blockDim.x + threadIdx.x;
    if (n < N_NODES) {
        float d = deg[n];
        dinv[n] = d > 0.f ? rsqrtf(d) : 0.f;
    }
}

__global__ void k_norm(const int* __restrict__ ei, const float* __restrict__ ew,
                       const float* __restrict__ dinv, float* __restrict__ nrm) {
    int stride = gridDim.x * blockDim.x;
    for (int e = blockIdx.x * blockDim.x + threadIdx.x; e < E_EDGES; e += stride) {
        int r = ei[e], c = ei[E_EDGES + e];
        nrm[e] = dinv[r] * ew[e] * dinv[c];
    }
}

// ---------------- WMMA GEMM: Y[N,128] = Abf[N,K] (bf16) @ W[K,128] ----------------
// Wt is W transposed: [H_DIM, K] bf16 so a lane's k-values lie in two contiguous 8-element runs.
// grid.x = N/32; block = 256 (8 waves); wave w owns output cols [16w,16w+16), 2 M-tiles.
template <int K>
__global__ __launch_bounds__(256) void k_gemm_wmma(const unsigned short* __restrict__ Abf,
                                                   const unsigned short* __restrict__ Wt,
                                                   float* __restrict__ Y) {
    constexpr int P  = K + 8;                 // LDS pitch (pad 4 DWORDs/row: 16B-aligned rows)
    constexpr int PI = (K == 256) ? 6 : 5;    // TDM pad_interval: 1<<PI 8-byte units == one row
    __shared__ unsigned short sA[32 * P];

    const int tid  = threadIdx.x;
    const int wave = tid >> 5;
    const int m0   = blockIdx.x * 32;

#if __has_builtin(__builtin_amdgcn_tensor_load_to_lds)
    if (wave == 0) {
        // TDM: 2D tile copy, 32 rows x K bf16, global row stride K, LDS pitch K+8.
        unsigned long long ga = (unsigned long long)(uintptr_t)(Abf + (size_t)m0 * K);
        unsigned ldsa = (unsigned)(uintptr_t)(void*)&sA[0];
        u32x4 g0;
        g0[0] = 1u;                                              // count=1 (valid user D#)
        g0[1] = ldsa;                                            // lds_addr
        g0[2] = (unsigned)(ga & 0xFFFFFFFFu);                    // global_addr[31:0]
        g0[3] = (unsigned)((ga >> 32) & 0x1FFFFFFu) | (2u << 30); // global_addr[56:32] | type=2
        const unsigned td0 = (unsigned)K;     // tensor_dim0 (elements)
        const unsigned td1 = 32u;             // tensor_dim1 (rows) == tile rows
        i32x8 g1;
        g1[0] = (int)((1u << 16)              // data_size = 1 (2-byte elements)
                    | (1u << 20)              // pad_enable
                    | ((unsigned)PI << 22)    // pad_interval = one row
                    | (3u << 25));            // pad_amount = 4 DWORDs (8 bf16)
        g1[1] = (int)((td0 & 0xFFFFu) << 16);                       // tensor_dim0[15:0]
        g1[2] = (int)(((td0 >> 16) & 0xFFFFu) | ((td1 & 0xFFFFu) << 16));
        g1[3] = (int)(((td1 >> 16) & 0xFFFFu) | ((unsigned)K << 16)); // tile_dim0 = K
        g1[4] = (int)32;                                            // tile_dim1 = 32
        g1[5] = (int)K;                                             // tensor_dim0_stride lo
        g1[6] = 0;
        g1[7] = 0;
        i32x4 gz4 = {0, 0, 0, 0};
        i32x8 gz8 = {0, 0, 0, 0, 0, 0, 0, 0};
        __builtin_amdgcn_tensor_load_to_lds(g0, g1, gz4, gz4, gz8, 0);
#if __has_builtin(__builtin_amdgcn_s_wait_tensorcnt)
        __builtin_amdgcn_s_wait_tensorcnt((short)0);
#else
        asm volatile("s_wait_tensorcnt 0x0" ::: "memory");
#endif
    }
#else
    for (int i = tid; i < 32 * K; i += 256) {
        int r = i / K, k = i - r * K;
        sA[r * P + k] = Abf[(m0 + r) * K + k];
    }
#endif
    __syncthreads();

    const int lane = tid & 31;
    const int n0   = wave * 16;
    const int m    = lane & 15;
    const int hi   = lane >> 4;
    const int colB = n0 + m;

    v8f c0 = {0.f, 0.f, 0.f, 0.f, 0.f, 0.f, 0.f, 0.f};
    v8f c1 = c0;

    #pragma unroll
    for (int k0 = 0; k0 < K; k0 += 32) {
        // B frag: lanes 0-15 hold K={0..7,16..23} of col N=lane; lanes 16-31 K={8..15,24..31}
        const unsigned short* wp = Wt + (size_t)colB * K + k0 + hi * 8;
        u16x8 blo = *(const u16x8*)wp;
        u16x8 bhi = *(const u16x8*)(wp + 16);
        v16u  bu  = __builtin_shufflevector(blo, bhi, 0, 1, 2, 3, 4, 5, 6, 7,
                                            8, 9, 10, 11, 12, 13, 14, 15);
        v16bf b = __builtin_bit_cast(v16bf, bu);

        // A frags (one per M-tile): lanes 0-15 rows M, K={0..7,16..23}; lanes 16-31 K={8..15,24..31}
        u16x8 a0lo = *(const u16x8*)&sA[(0 + m) * P + k0 + hi * 8];
        u16x8 a0hi = *(const u16x8*)&sA[(0 + m) * P + k0 + 16 + hi * 8];
        v16u  a0u  = __builtin_shufflevector(a0lo, a0hi, 0, 1, 2, 3, 4, 5, 6, 7,
                                             8, 9, 10, 11, 12, 13, 14, 15);
        u16x8 a1lo = *(const u16x8*)&sA[(16 + m) * P + k0 + hi * 8];
        u16x8 a1hi = *(const u16x8*)&sA[(16 + m) * P + k0 + 16 + hi * 8];
        v16u  a1u  = __builtin_shufflevector(a1lo, a1hi, 0, 1, 2, 3, 4, 5, 6, 7,
                                             8, 9, 10, 11, 12, 13, 14, 15);
        v16bf a0 = __builtin_bit_cast(v16bf, a0u);
        v16bf a1 = __builtin_bit_cast(v16bf, a1u);

        c0 = __builtin_amdgcn_wmma_f32_16x16x32_bf16(false, a0, false, b, (short)0, c0, false, false);
        c1 = __builtin_amdgcn_wmma_f32_16x16x32_bf16(false, a1, false, b, (short)0, c1, false, false);
    }

    #pragma unroll
    for (int r = 0; r < 8; ++r) {           // C/D layout: VGPR r = M=r (lanes<16) / M=8+r
        int row0 = m0 + hi * 8 + r;
        int row1 = m0 + 16 + hi * 8 + r;
        Y[row0 * H_DIM + n0 + m] = c0[r];
        Y[row1 * H_DIM + n0 + m] = c1[r];
    }
}

// ---------------- aggregation ----------------
__global__ void k_agg_init(const float* __restrict__ xw, const float* __restrict__ dinv,
                           const float* __restrict__ bias, float* __restrict__ agg) {
    int i = blockIdx.x * blockDim.x + threadIdx.x;
    if (i < N_NODES * H_DIM) {
        int n = i >> 7, ch = i & 127;
        float dn = dinv[n];
        agg[i] = dn * dn * xw[i] + bias[ch];
    }
}

__global__ __launch_bounds__(256) void k_agg_edges(const int* __restrict__ ei,
                                                   const float* __restrict__ nrm,
                                                   const float* __restrict__ xw,
                                                   float* __restrict__ agg) {
    int ch  = threadIdx.x & 127;
    int sub = threadIdx.x >> 7;                      // 2 edges per block-iteration
    int stride = gridDim.x * 2;
    for (int e = blockIdx.x * 2 + sub; e < E_EDGES; e += stride) {
        __builtin_prefetch(&ei[e + stride], 0, 0);
        __builtin_prefetch(&ei[E_EDGES + e + stride], 0, 0);
        __builtin_prefetch(&nrm[e + stride], 0, 0);
        int r = ei[e], c = ei[E_EDGES + e];
        float w = nrm[e];
        atomAddF32(&agg[c * H_DIM + ch], w * xw[r * H_DIM + ch]);
    }
}

// ---------------- batchnorm ----------------
__global__ __launch_bounds__(128) void k_bn_stats(const float* __restrict__ h,
                                                  float* __restrict__ sum,
                                                  float* __restrict__ sq) {
    int ch = threadIdx.x;                            // 128 threads = 128 channels
    int n0 = blockIdx.x * 256;
    int n1 = n0 + 256; if (n1 > N_NODES) n1 = N_NODES;
    float s = 0.f, q = 0.f;
    for (int n = n0; n < n1; ++n) {
        float v = h[n * H_DIM + ch];
        s += v; q += v * v;
    }
    atomAddF32(&sum[ch], s);
    atomAddF32(&sq[ch], q);
}

// BN + ReLU; writes f32 (for logits / aggregation) and bf16 (next layer's GEMM A operand)
__global__ void k_bn_relu(float* __restrict__ hio, unsigned short* __restrict__ hbf,
                          const float* __restrict__ sum, const float* __restrict__ sq,
                          const float* __restrict__ g, const float* __restrict__ beta) {
    int i = blockIdx.x * blockDim.x + threadIdx.x;
    if (i < N_NODES * H_DIM) {
        int ch = i & 127;
        const float invN = 1.0f / (float)N_NODES;
        float mean = sum[ch] * invN;
        float var  = sq[ch] * invN - mean * mean;
        var = var > 0.f ? var : 0.f;
        float v = (hio[i] - mean) * rsqrtf(var + BN_EPS) * g[ch] + beta[ch];
        v = v > 0.f ? v : 0.f;
        hio[i] = v;
        hbf[i] = f2bf(v);
    }
}

// ---------------- logits + softmax ----------------
__global__ __launch_bounds__(256) void k_logits_softmax(const float* __restrict__ h,
                                                        const float* __restrict__ Wf,
                                                        const float* __restrict__ bf,
                                                        float* __restrict__ out) {
    int n = blockIdx.x * 16 + (threadIdx.x >> 4);
    int c = threadIdx.x & 15;
    if (n >= N_NODES) return;
    float acc = bf[c];
    for (int ch = 0; ch < H_DIM; ++ch)
        acc += h[n * H_DIM + ch] * Wf[ch * C_OUT + c];
    float mx = acc;
    #pragma unroll
    for (int off = 8; off > 0; off >>= 1)
        mx = fmaxf(mx, __shfl_xor(mx, off, 16));
    float e = __expf(acc - mx);
    float s = e;
    #pragma unroll
    for (int off = 8; off > 0; off >>= 1)
        s += __shfl_xor(s, off, 16);
    out[n * C_OUT + c] = e / s;
}

// ---------------- launcher ----------------
extern "C" void kernel_launch(void* const* d_in, const int* in_sizes, int n_in,
                              void* d_out, int out_size, void* d_ws, size_t ws_size,
                              hipStream_t stream) {
    const float* x   = (const float*)d_in[0];
    const int*   ei  = (const int*)  d_in[1];
    const float* ew  = (const float*)d_in[2];
    const float* W1  = (const float*)d_in[3];
    const float* b1  = (const float*)d_in[4];
    const float* W2  = (const float*)d_in[5];
    const float* b2  = (const float*)d_in[6];
    const float* W3  = (const float*)d_in[7];
    const float* b3  = (const float*)d_in[8];
    const float* g1  = (const float*)d_in[9];
    const float* be1 = (const float*)d_in[10];
    const float* g2  = (const float*)d_in[11];
    const float* be2 = (const float*)d_in[12];
    const float* g3  = (const float*)d_in[13];
    const float* be3 = (const float*)d_in[14];
    const float* Wf  = (const float*)d_in[15];
    const float* bfv = (const float*)d_in[16];
    float* out = (float*)d_out;

    char* ws = (char*)d_ws;
    float*          xw   = (float*)(ws);                          // N*H f32    (51.2 MB)
    float*          hbuf = (float*)(ws + 51200000);               // N*H f32    (51.2 MB)
    unsigned short* hbf  = (unsigned short*)(ws + 102400000);     // N*H bf16   (25.6 MB)
    unsigned short* xbf  = (unsigned short*)(ws + 128000000);     // N*F_IN bf16(51.2 MB)
    float*          nrm  = (float*)(ws + 179200000);              // E f32      ( 6.4 MB)
    float*          deg  = (float*)(ws + 185600000);              // N f32
    float*          dinv = (float*)(ws + 186000000);              // N f32
    float*          bsum = (float*)(ws + 186400000);              // H f32
    float*          bsq  = (float*)(ws + 186400512);              // H f32 (contiguous)
    unsigned short* W1t  = (unsigned short*)(ws + 186401024);     // [128,256] bf16
    unsigned short* W2t  = (unsigned short*)(ws + 186466560);     // [128,128] bf16
    unsigned short* W3t  = (unsigned short*)(ws + 186499328);     // [128,128] bf16

    const int TB  = 256;
    const int gN  = (N_NODES + TB - 1) / TB;
    const int gNH = (N_NODES * H_DIM + TB - 1) / TB;

    // gcn_norm
    k_fill<<<gN, TB, 0, stream>>>(deg, 1.0f, N_NODES);            // self-loop weight
    k_deg<<<4096, TB, 0, stream>>>(ei, ew, deg);
    k_dinv<<<gN, TB, 0, stream>>>(deg, dinv);
    k_norm<<<4096, TB, 0, stream>>>(ei, ew, dinv, nrm);

    // bf16 operands: x once, weights transposed once
    k_cvt_bf16<<<(N_NODES * F_IN + TB - 1) / TB, TB, 0, stream>>>(x, xbf, N_NODES * F_IN);
    k_cvt_tr_bf16<<<(F_IN * H_DIM + TB - 1) / TB, TB, 0, stream>>>(W1, W1t, F_IN);
    k_cvt_tr_bf16<<<(H_DIM * H_DIM + TB - 1) / TB, TB, 0, stream>>>(W2, W2t, H_DIM);
    k_cvt_tr_bf16<<<(H_DIM * H_DIM + TB - 1) / TB, TB, 0, stream>>>(W3, W3t, H_DIM);

    auto post_gemm = [&](const float* bias, const float* gamma, const float* beta) {
        k_agg_init<<<gNH, TB, 0, stream>>>(xw, dinv, bias, hbuf);
        k_agg_edges<<<8192, 256, 0, stream>>>(ei, nrm, xw, hbuf);
        k_fill<<<1, 256, 0, stream>>>(bsum, 0.f, 2 * H_DIM);      // zero sum+sumsq
        k_bn_stats<<<(N_NODES + 255) / 256, 128, 0, stream>>>(hbuf, bsum, bsq);
        k_bn_relu<<<gNH, TB, 0, stream>>>(hbuf, hbf, bsum, bsq, gamma, beta);
    };

    k_gemm_wmma<F_IN><<<N_NODES / 32, 256, 0, stream>>>(xbf, W1t, xw);
    post_gemm(b1, g1, be1);
    k_gemm_wmma<H_DIM><<<N_NODES / 32, 256, 0, stream>>>(hbf, W2t, xw);
    post_gemm(b2, g2, be2);
    k_gemm_wmma<H_DIM><<<N_NODES / 32, 256, 0, stream>>>(hbf, W3t, xw);
    post_gemm(b3, g3, be3);

    k_logits_softmax<<<(N_NODES + 15) / 16, 256, 0, stream>>>(hbuf, Wf, bfv, out);
}
